// LocalAttentionBlock_32478542692694
// MI455X (gfx1250) — matrix operand order repaired and checked by reading, not compiled, not run on hardware
//
#include <hip/hip_runtime.h>
#include <hip/hip_bf16.h>
#include <math.h>

typedef _Float16 half_t;
typedef __attribute__((ext_vector_type(16))) _Float16 v16h;
typedef __attribute__((ext_vector_type(8)))  float    v8f;
typedef int v4i_ __attribute__((vector_size(16)));

#define DIM 256
#define HEADS 8
#define HD 32
#define WIN 8
#define T 64           // tokens per window
#define NWIN 4096      // B * (H/WIN) * (W/WIN) = 4*32*32

// ---------------------------------------------------------------------------
// Address-space helpers for async global->LDS copies (gfx1250 ASYNCcnt path).
// Generic LDS pointers carry the LDS offset in the low 32 bits; generic global
// pointers are the global address.
// ---------------------------------------------------------------------------
#define AS3 __attribute__((address_space(3)))
#define AS1 __attribute__((address_space(1)))

#if __has_builtin(__builtin_amdgcn_global_load_async_to_lds_b128)
#define HAS_ASYNC_LDS 1
#else
#define HAS_ASYNC_LDS 0
#endif

__device__ __forceinline__ void async_copy_b128(half_t* lds_dst,
                                                const half_t* gsrc) {
#if HAS_ASYNC_LDS
  __builtin_amdgcn_global_load_async_to_lds_b128(
      (AS1 v4i_*)(uintptr_t)gsrc,
      (AS3 v4i_*)(uint32_t)(uintptr_t)lds_dst,
      0, 0);
#else
  *(uint4*)lds_dst = *(const uint4*)gsrc;
#endif
}

__device__ __forceinline__ void async_copy_wait() {
#if HAS_ASYNC_LDS
#if __has_builtin(__builtin_amdgcn_s_wait_asynccnt)
  __builtin_amdgcn_s_wait_asynccnt(0);
#else
  asm volatile("s_wait_asynccnt 0x0" ::: "memory");
#endif
#endif
}

// ---------------------------------------------------------------------------
// WMMA fragment helpers (CDNA5 wave32 layouts, CDNA5 ISA 7.12.2)
// ---------------------------------------------------------------------------

// A-matrix 16x32 f16: lane<16 -> M=lane, K={kb..kb+7, kb+16..kb+23} with kb=0;
// lane>=16 -> M=lane-16, kb=8.
__device__ __forceinline__ v16h load_a_frag(const half_t* A, int row0, int k0,
                                            int lda, int lane) {
  int row = row0 + (lane & 15);
  int kb  = k0 + ((lane >> 4) << 3);
  const half_t* p0 = A + row * lda + kb;
  const half_t* p1 = p0 + 16;
  v16h a;
#pragma unroll
  for (int j = 0; j < 8; ++j) { a[j] = p0[j]; a[8 + j] = p1[j]; }
  return a;
}

// B-matrix 32x16 f16: lane holds column N = n0+(lane&15), K = {kb..kb+7, kb+16..kb+23}, kb = 8*(lane>>4).
__device__ __forceinline__ v16h load_b_frag(const half_t* Bm, int k0, int n0,
                                            int ldb, int lane) {
  int kb = k0 + ((lane >> 4) << 3);
  const half_t* p = Bm + (size_t)kb * ldb + n0 + (lane & 15);
  v16h b;
#pragma unroll
  for (int j = 0; j < 8; ++j) { b[j] = p[(size_t)j * ldb]; b[8 + j] = p[(size_t)(16 + j) * ldb]; }
  return b;
}

__device__ __forceinline__ v8f wmma_f16(v16h a, v16h b, v8f c) {
  return __builtin_amdgcn_wmma_f32_16x16x32_f16(
      /*neg_a=*/false, a, /*neg_b=*/false, b,
      /*c_mod=*/(short)0, c, /*reuse_a=*/false, /*reuse_b=*/false);
}

// ---------------------------------------------------------------------------
// Kernel 0: fp32 weights -> K-major f16 (B-matrix friendly)
//   Bin [k][e] = w_in[e][k]   (256 x 768)
//   Bout[k][o] = w_out[o][k]  (256 x 256)
// ---------------------------------------------------------------------------
__global__ __launch_bounds__(256) void wprep_kernel(const float* __restrict__ w_in,
                                                    const float* __restrict__ w_out,
                                                    half_t* __restrict__ Bin,
                                                    half_t* __restrict__ Bout) {
  int idx = blockIdx.x * 256 + threadIdx.x;   // grid: 768 blocks -> 196608
  if (idx < 768 * 256) {
    int k = idx / 768, e = idx % 768;
    Bin[idx] = (half_t)w_in[e * 256 + k];
  }
  if (idx < 256 * 256) {
    int k = idx >> 8, o = idx & 255;
    Bout[idx] = (half_t)w_out[o * 256 + k];
  }
}

// ---------------------------------------------------------------------------
// Kernel 1: window gather + fused QKV projection.
// One block (256 thr = 8 waves) per window. M=64, N=768, K=256.
//   Q -> [n][h][t][d], K -> [n][h][d][t] (pre-transposed), V -> [n][h][t][d]
// ---------------------------------------------------------------------------
#define XLD 264
__global__ __launch_bounds__(256) void qkv_kernel(const float* __restrict__ x,
                                                  const half_t* __restrict__ Bin,
                                                  const float* __restrict__ b_in,
                                                  half_t* __restrict__ Qws,
                                                  half_t* __restrict__ Kws,
                                                  half_t* __restrict__ Vws) {
  __shared__ half_t Xs[T * XLD];   // xw[t][c] f16, ~33.8 KB

  int n = blockIdx.x;
  int b  = n >> 10;
  int rem = n & 1023;
  int h0 = (rem >> 5) << 3;
  int w0 = (rem & 31) << 3;
  int tid = threadIdx.x;

  // Gather: thread tid = channel c; 8x8 patch -> column c of Xs.
  {
    int c = tid;
    const float* xp = x + (((size_t)(b * DIM + c) * 256 + h0) * 256 + w0);
#pragma unroll
    for (int s1 = 0; s1 < 8; ++s1) {
#pragma unroll
      for (int s2 = 0; s2 < 8; ++s2) {
        Xs[(s1 * 8 + s2) * XLD + c] = (half_t)xp[s1 * 256 + s2];
      }
    }
  }
  __syncthreads();

  int wave = tid >> 5, lane = tid & 31;

  for (int i = 0; i < 6; ++i) {
    int nt = wave * 6 + i;          // 48 column tiles of 16
    int e0 = nt * 16;
    __builtin_prefetch(Bin + (size_t)e0 + 16, 0, 1);
    int col   = e0 + (lane & 15);
    float bias = b_in[col];
#pragma unroll
    for (int mt = 0; mt < 4; ++mt) {
      v8f acc = {};
#pragma unroll
      for (int kk = 0; kk < 8; ++kk) {
        v16h a  = load_a_frag(Xs, mt * 16, kk * 32, XLD, lane);
        v16h bb = load_b_frag(Bin, kk * 32, e0, 768, lane);
        acc = wmma_f16(a, bb, acc);
      }
      int rbase = mt * 16 + ((lane >> 4) << 3);
      // Branch on tile-uniform e0 (not lane-dependent col): one branch per
      // tile instead of per-element exec-masked stores.
      if (e0 < 256) {               // Q
        int h = col >> 5, d = col & 31;
        half_t* qp = Qws + (((size_t)n * HEADS + h) * T + rbase) * HD + d;
#pragma unroll
        for (int r = 0; r < 8; ++r) qp[r * HD] = (half_t)(acc[r] + bias);
      } else if (e0 < 512) {        // K (store transposed: [d][t])
        int cc = col - 256, h = cc >> 5, d = cc & 31;
        half_t* kp = Kws + (((size_t)n * HEADS + h) * HD + d) * T + rbase;
#pragma unroll
        for (int r = 0; r < 8; ++r) kp[r] = (half_t)(acc[r] + bias);
      } else {                      // V
        int cc = col - 512, h = cc >> 5, d = cc & 31;
        half_t* vp = Vws + (((size_t)n * HEADS + h) * T + rbase) * HD + d;
#pragma unroll
        for (int r = 0; r < 8; ++r) vp[r * HD] = (half_t)(acc[r] + bias);
      }
    }
  }
}

// ---------------------------------------------------------------------------
// Kernel 2: per-(window,head) attention. Block = 128 thr = 4 waves.
// Q/K^T/V (12 KB) are async-copied into LDS (ASYNCcnt path), then
// S = Q K^T * scale -> softmax -> O = P V, all WMMA fed from LDS.
// O -> [n][t][h*32+d] (row-major A for the out-proj GEMM).
// ---------------------------------------------------------------------------
__global__ __launch_bounds__(128) void attn_kernel(const half_t* __restrict__ Qws,
                                                   const half_t* __restrict__ Kws,
                                                   const half_t* __restrict__ Vws,
                                                   half_t* __restrict__ Ows) {
  __shared__ half_t Qs[T * HD];   // [64][32]  4 KB
  __shared__ half_t Ks[HD * T];   // [32][64]  4 KB (= K^T)
  __shared__ half_t Vs[T * HD];   // [64][32]  4 KB
  __shared__ float  Ss[T * 66];   // scores / exp, 16.9 KB
  __shared__ half_t Ps[T * 72];   // probs f16, 9.2 KB

  int nh = blockIdx.x;
  int n = nh >> 3, h = nh & 7;
  const half_t* Q = Qws + ((size_t)n * HEADS + h) * T * HD;
  const half_t* K = Kws + ((size_t)n * HEADS + h) * HD * T;
  const half_t* V = Vws + ((size_t)n * HEADS + h) * T * HD;

  int tid = threadIdx.x, wave = tid >> 5, lane = tid & 31;
  const float scale = 0.17677669529663687f;   // 1/sqrt(32)

  // Async-stage Q, K^T, V into LDS: 3 * 256 b128-chunks over 128 threads.
#pragma unroll
  for (int r = 0; r < 2; ++r) {
    int idx = tid + r * 128;        // chunk id, 16 B each
    async_copy_b128(Qs + idx * 8, Q + idx * 8);
    async_copy_b128(Ks + idx * 8, K + idx * 8);
    async_copy_b128(Vs + idx * 8, V + idx * 8);
  }
  async_copy_wait();
  __syncthreads();

  // S = Q K^T : 16 tiles (4x4), wave handles row mt=wave, all 4 col tiles.
#pragma unroll
  for (int i = 0; i < 4; ++i) {
    v16h a  = load_a_frag(Qs, wave * 16, 0, HD, lane);
    v16h bb = load_b_frag(Ks, 0, i * 16, T, lane);
    v8f acc = {};
    acc = wmma_f16(a, bb, acc);
    int col = i * 16 + (lane & 15);
    int rbase = wave * 16 + ((lane >> 4) << 3);
#pragma unroll
    for (int r = 0; r < 8; ++r) Ss[(rbase + r) * 66 + col] = acc[r] * scale;
  }
  __syncthreads();

  // Softmax: one thread per row.
  if (tid < T) {
    float* row = Ss + tid * 66;
    float m = row[0];
    for (int j = 1; j < T; ++j) m = fmaxf(m, row[j]);
    float s = 0.f;
    for (int j = 0; j < T; ++j) { float e = __expf(row[j] - m); row[j] = e; s += e; }
    float inv = 1.f / s;
    half_t* prow = Ps + tid * 72;
    for (int j = 0; j < T; ++j) prow[j] = (half_t)(row[j] * inv);
  }
  __syncthreads();

  // O = P V : M=64, N=32, K=64 -> 8 tiles, 2 per wave, 2 k-steps each.
#pragma unroll
  for (int i = 0; i < 2; ++i) {
    int id = wave * 2 + i;
    int mt = id >> 1, ntile = id & 1;
    v8f acc = {};
#pragma unroll
    for (int kk = 0; kk < 2; ++kk) {
      v16h a  = load_a_frag(Ps, mt * 16, kk * 32, 72, lane);
      v16h bb = load_b_frag(Vs, kk * 32, ntile * 16, HD, lane);
      acc = wmma_f16(a, bb, acc);
    }
    int col = ntile * 16 + (lane & 15);
    int rbase = mt * 16 + ((lane >> 4) << 3);
    half_t* op = Ows + ((size_t)n * T + rbase) * DIM + h * HD + col;
#pragma unroll
    for (int r = 0; r < 8; ++r) op[r * DIM] = (half_t)acc[r];
  }
}

// ---------------------------------------------------------------------------
// Kernel 3: out-proj GEMM + channel LayerNorm + scatter to NCHW fp32 output.
// One block (256 thr = 8 waves) per window. M=64, N=256, K=256.
// ---------------------------------------------------------------------------
__global__ __launch_bounds__(256) void outln_kernel(const half_t* __restrict__ Ows,
                                                    const half_t* __restrict__ Bout,
                                                    const float* __restrict__ b_out,
                                                    const float* __restrict__ gamma,
                                                    const float* __restrict__ beta,
                                                    float* __restrict__ out) {
  __shared__ half_t Ys[T * XLD];   // projected rows, ~33.8 KB

  int n = blockIdx.x;
  const half_t* A = Ows + (size_t)n * T * DIM;   // [64][256] row-major
  int tid = threadIdx.x, wave = tid >> 5, lane = tid & 31;

  for (int i = 0; i < 8; ++i) {
    int id = wave * 8 + i;        // 64 tiles = 4 mt x 16 ntile
    int ntile = id >> 2, mt = id & 3;
    v8f acc = {};
#pragma unroll
    for (int kk = 0; kk < 8; ++kk) {
      v16h a  = load_a_frag(A, mt * 16, kk * 32, DIM, lane);
      v16h bb = load_b_frag(Bout, kk * 32, ntile * 16, DIM, lane);
      acc = wmma_f16(a, bb, acc);
    }
    int col = ntile * 16 + (lane & 15);
    int rbase = mt * 16 + ((lane >> 4) << 3);
    float bias = b_out[col];
#pragma unroll
    for (int r = 0; r < 8; ++r)
      Ys[(rbase + r) * XLD + col] = (half_t)(acc[r] + bias);
  }
  __syncthreads();

  // LayerNorm over channels per token, scatter back to [B][C][H][W].
  int b = n >> 10;
  int rem = n & 1023;
  int h0 = (rem >> 5) << 3;
  int w0 = (rem & 31) << 3;
  if (tid < T) {
    int t = tid, s1 = t >> 3, s2 = t & 7;
    const half_t* yr = Ys + t * XLD;
    float mean = 0.f;
    for (int c = 0; c < DIM; ++c) mean += (float)yr[c];
    mean *= (1.0f / DIM);
    float var = 0.f;
    for (int c = 0; c < DIM; ++c) { float d = (float)yr[c] - mean; var += d * d; }
    var *= (1.0f / DIM);
    float inv = rsqrtf(var + 1e-5f);
    size_t base = (size_t)b * DIM * 65536 + (size_t)(h0 + s1) * 256 + (w0 + s2);
    for (int c = 0; c < DIM; ++c) {
      out[base + (size_t)c * 65536] =
          ((float)yr[c] - mean) * inv * gamma[c] + beta[c];
    }
  }
}

// ---------------------------------------------------------------------------
extern "C" void kernel_launch(void* const* d_in, const int* in_sizes, int n_in,
                              void* d_out, int out_size, void* d_ws, size_t ws_size,
                              hipStream_t stream) {
  const float* x     = (const float*)d_in[0];
  const float* w_in  = (const float*)d_in[1];
  const float* b_in  = (const float*)d_in[2];
  const float* w_out = (const float*)d_in[3];
  const float* b_out = (const float*)d_in[4];
  const float* gamma = (const float*)d_in[5];
  const float* beta  = (const float*)d_in[6];
  float* out = (float*)d_out;

  char* ws = (char*)d_ws;
  size_t off = 0;
  half_t* Bin  = (half_t*)(ws + off); off += (size_t)768 * 256 * sizeof(half_t);
  half_t* Bout = (half_t*)(ws + off); off += (size_t)256 * 256 * sizeof(half_t);
  off = (off + 255) & ~(size_t)255;
  size_t qkv_elems = (size_t)NWIN * HEADS * T * HD;       // 67,108,864
  half_t* Qws = (half_t*)(ws + off); off += qkv_elems * sizeof(half_t);
  half_t* Kws = (half_t*)(ws + off); off += qkv_elems * sizeof(half_t);
  half_t* Vws = (half_t*)(ws + off); off += qkv_elems * sizeof(half_t);
  half_t* Ows = (half_t*)(ws + off); off += (size_t)NWIN * T * DIM * sizeof(half_t);

  wprep_kernel<<<768, 256, 0, stream>>>(w_in, w_out, Bin, Bout);
  qkv_kernel<<<NWIN, 256, 0, stream>>>(x, Bin, b_in, Qws, Kws, Vws);
  attn_kernel<<<NWIN * HEADS, 128, 0, stream>>>(Qws, Kws, Vws, Ows);
  outln_kernel<<<NWIN, 256, 0, stream>>>(Ows, Bout, b_out, gamma, beta, out);
}
